// STGCN_23227183136950
// MI455X (gfx1250) — compile-verified, matrix operand order checked by reading, not run on hardware
//
#include <hip/hip_runtime.h>
#include <hip/hip_bf16.h>

// ---------------------------------------------------------------------------
// ST-GCN forward for MI455X (gfx1250).
// All conv GEMMs run on v_wmma_f32_16x16x32_bf16 (fp32 accumulate).
// Activation layout (fp32 block in/out): channel-major act[c][n], n=(b*T+t)*V+v.
// GCN->TCN intermediate y: bf16 n-major y[n][C] (vector-friendly implicit conv).
// TCN B-staging uses gfx1250 async global->LDS copies (ASYNCcnt).
//
// WMMA operands pre-swizzled in global memory into the ISA fragment layout:
// per (tile16 x ktile32) block of 512 bf16 = [2 halves][32 lanes][8]:
//   lane l -> row/col = (l&15), hs = l>>4
//   half h -> K = kt*32 + h*16 + hs*8 + {0..7}
// so a fragment is exactly two global_load_b128 per operand.
// ---------------------------------------------------------------------------

typedef __bf16 v16bf __attribute__((ext_vector_type(16)));
typedef __bf16 v8bfv __attribute__((ext_vector_type(8)));
typedef float  v8f   __attribute__((ext_vector_type(8)));
typedef unsigned short v8us __attribute__((ext_vector_type(8)));

__device__ __forceinline__ unsigned short f2bf(float f) {
    unsigned int u = __builtin_bit_cast(unsigned int, f);
    unsigned int r = u + 0x7FFFu + ((u >> 16) & 1u);   // round-to-nearest-even
    return (unsigned short)(r >> 16);
}

__device__ __forceinline__ v16bf frag16(v8bfv lo, v8bfv hi) {
    v16bf a;
#pragma unroll
    for (int i = 0; i < 8; i++) { a[i] = lo[i]; a[8 + i] = hi[i]; }
    return a;
}

// Epilogue value chain: +bias -> BN -> +res -> ReLU -> BN2+ReLU (each optional).
__device__ __forceinline__ float ep_val(
    float v, int m, int n,
    const float* __restrict__ bias, int biasMode, int M,
    const float* __restrict__ bnG, const float* __restrict__ bnB,
    const float* __restrict__ bnM, const float* __restrict__ bnV,
    const float* __restrict__ resPtr, int doRelu,
    const float* __restrict__ bn2G, const float* __restrict__ bn2B,
    const float* __restrict__ bn2M, const float* __restrict__ bn2V,
    int Nout)
{
    const float eps = 1e-5f;
    if (biasMode == 1)      v += bias[m];
    else if (biasMode == 2) v += bias[m] + bias[M + m] + bias[2 * M + m];
    if (bnG) v = (v - bnM[m]) * (bnG[m] * rsqrtf(bnV[m] + eps)) + bnB[m];
    if (resPtr) v += resPtr[(size_t)m * Nout + n];
    if (doRelu) v = fmaxf(v, 0.0f);
    if (bn2G) {
        v = (v - bn2M[m]) * (bn2G[m] * rsqrtf(bn2V[m] + eps)) + bn2B[m];
        v = fmaxf(v, 0.0f);
    }
    return v;
}

// ---------------------------------------------------------------------------
// GEMM 1 (GCN main): both operands pre-swizzled bf16 in global memory.
// No LDS, no barriers, no bounds checks (M%64==0, Nout%128==0, K padded to 32).
// Block tile 64M x 128N, 8 waves (4x2), wave tile 16M x 64N -> 4 wmma/K-step.
// Output: bf16 n-major Out[n][M] via one global_store_b128 per tile per lane.
// ---------------------------------------------------------------------------
__launch_bounds__(256)
__global__ void wmma_gemm_swz_kernel(
    const unsigned short* __restrict__ Aswz,
    const unsigned short* __restrict__ Bswz,
    unsigned short* __restrict__ OutBF,              // [Nout][M] bf16
    const float* __restrict__ bias, int biasMode,
    const float* __restrict__ bnG, const float* __restrict__ bnB,
    const float* __restrict__ bnM, const float* __restrict__ bnV,
    const float* __restrict__ resPtr, int doRelu,
    const float* __restrict__ bn2G, const float* __restrict__ bn2B,
    const float* __restrict__ bn2M, const float* __restrict__ bn2V,
    int M, int Nout, int numKT)
{
    const int tid  = threadIdx.x;
    const int lane = tid & 31;
    const int wave = tid >> 5;
    const int wm   = wave & 3;
    const int wn   = wave >> 2;
    const int lr   = lane & 15;
    const int hs   = lane >> 4;

    const int mt     = blockIdx.y * 4 + wm;
    const int ntBase = blockIdx.x * 8 + wn * 4;

    v8f acc[4];
#pragma unroll
    for (int j = 0; j < 4; j++)
#pragma unroll
        for (int e = 0; e < 8; e++) acc[j][e] = 0.0f;

    const unsigned short* aBase = Aswz + ((size_t)mt * numKT) * 512 + (size_t)lane * 8;

    for (int kt = 0; kt < numKT; kt++) {
        v8bfv alo = *reinterpret_cast<const v8bfv*>(aBase + (size_t)kt * 512);
        v8bfv ahi = *reinterpret_cast<const v8bfv*>(aBase + (size_t)kt * 512 + 256);
        v16bf a = frag16(alo, ahi);
#pragma unroll
        for (int j = 0; j < 4; j++) {
            const unsigned short* bBase =
                Bswz + ((size_t)(ntBase + j) * numKT + kt) * 512 + (size_t)lane * 8;
            v8bfv blo = *reinterpret_cast<const v8bfv*>(bBase);
            v8bfv bhi = *reinterpret_cast<const v8bfv*>(bBase + 256);
            v16bf b = frag16(blo, bhi);
            acc[j] = __builtin_amdgcn_wmma_f32_16x16x32_bf16(
                false, a, false, b, (short)0, acc[j], false, false);
        }
    }

    // epilogue: lane's 8 rows are contiguous channels in [n][M] layout
    const int mRow = mt * 16 + 8 * hs;
#pragma unroll
    for (int j = 0; j < 4; j++) {
        int n = (ntBase + j) * 16 + lr;
        v8us o;
#pragma unroll
        for (int r = 0; r < 8; r++) {
            float v = ep_val(acc[j][r], mRow + r, n, bias, biasMode, M,
                             bnG, bnB, bnM, bnV, resPtr, doRelu,
                             bn2G, bn2B, bn2M, bn2V, Nout);
            o[r] = f2bf(v);
        }
        *reinterpret_cast<v8us*>(OutBF + (size_t)n * M + mRow) = o;
    }
}

// ---------------------------------------------------------------------------
// GEMM 2 (1x1 residual convs, optional temporal stride): A pre-swizzled bf16;
// B gathered from fp32 [K][Nsrc] through LDS, branch-free, K-invariant
// n-decomposition hoisted out of the K loop. fp32 [M][Nout] output.
// ---------------------------------------------------------------------------
__launch_bounds__(256)
__global__ void wmma_gemm_1x1_kernel(
    const unsigned short* __restrict__ Aswz,
    const float* __restrict__ srcF,
    float* __restrict__ Out,
    const float* __restrict__ bias, int biasMode,
    const float* __restrict__ bnG, const float* __restrict__ bnB,
    const float* __restrict__ bnM, const float* __restrict__ bnV,
    int M, int K, int Nout, int Nsrc, int numKT,
    int Tin, int Tout, int stride)
{
    __shared__ unsigned short Bs[128][32];

    const int tid  = threadIdx.x;
    const int lane = tid & 31;
    const int wave = tid >> 5;
    const int wm   = wave & 3;
    const int wn   = wave >> 2;
    const int lr   = lane & 15;
    const int hs   = lane >> 4;

    const int n0 = blockIdx.x * 128;
    const int mt = blockIdx.y * 4 + wm;

    v8f acc[4];
#pragma unroll
    for (int j = 0; j < 4; j++)
#pragma unroll
        for (int e = 0; e < 8; e++) acc[j][e] = 0.0f;

    // hoisted K-invariant source row index per staging element
    int baseIdx[16];
#pragma unroll
    for (int i = 0; i < 16; i++) {
        int e  = tid + i * 256;
        int nl = e >> 5;
        int n  = n0 + nl;
        int vv = n % 17;
        int r2 = n / 17;
        int to = r2 % Tout;
        int b  = r2 / Tout;
        baseIdx[i] = (b * Tin + to * stride) * 17 + vv;
    }

    const unsigned short* aBase = Aswz + ((size_t)mt * numKT) * 512 + (size_t)lane * 8;

    for (int kt = 0; kt < numKT; kt++) {
        const int k0 = kt * 32;
#pragma unroll
        for (int i = 0; i < 16; i++) {
            int e  = tid + i * 256;
            int nl = e >> 5, c = e & 31;
            int k = k0 + c;
            bool ok = k < K;
            int kk = ok ? k : 0;
            float f = srcF[(size_t)kk * Nsrc + baseIdx[i]];
            Bs[nl][c] = ok ? f2bf(f) : (unsigned short)0;
        }
        __syncthreads();

        v8bfv alo = *reinterpret_cast<const v8bfv*>(aBase + (size_t)kt * 512);
        v8bfv ahi = *reinterpret_cast<const v8bfv*>(aBase + (size_t)kt * 512 + 256);
        v16bf a = frag16(alo, ahi);
#pragma unroll
        for (int j = 0; j < 4; j++) {
            int nb = wn * 64 + j * 16;
            v8bfv blo = *reinterpret_cast<const v8bfv*>(&Bs[nb + lr][hs * 8]);
            v8bfv bhi = *reinterpret_cast<const v8bfv*>(&Bs[nb + lr][16 + hs * 8]);
            v16bf b = frag16(blo, bhi);
            acc[j] = __builtin_amdgcn_wmma_f32_16x16x32_bf16(
                false, a, false, b, (short)0, acc[j], false, false);
        }
        __syncthreads();
    }

#pragma unroll
    for (int j = 0; j < 4; j++) {
#pragma unroll
        for (int r = 0; r < 8; r++) {
            int m = mt * 16 + r + 8 * hs;
            int n = n0 + wn * 64 + j * 16 + lr;
            float v = ep_val(acc[j][r], m, n, bias, biasMode, M,
                             bnG, bnB, bnM, bnV, nullptr, 0,
                             nullptr, nullptr, nullptr, nullptr, Nout);
            Out[(size_t)m * Nout + n] = v;
        }
    }
}

// ---------------------------------------------------------------------------
// GEMM 3 (TCN implicit 9-tap temporal conv, stride s): A pre-swizzled bf16;
// B from bf16 n-major y[n][C]. Since C%32==0, each 32-wide K-tile lies in a
// single tap dt. B-staging uses gfx1250 async global->LDS B128 copies
// (ASYNCcnt); conv-padding taps are redirected to a zero page (branch-free).
// fp32 [M][Nout] output with residual/BN/ReLU epilogue.
// ---------------------------------------------------------------------------
__launch_bounds__(256)
__global__ void wmma_gemm_tconv_kernel(
    const unsigned short* __restrict__ Aswz,
    const unsigned short* __restrict__ yBF,          // [Nin_pos][C] bf16
    const unsigned short* __restrict__ zeroPage,     // >=32B of zeros
    float* __restrict__ Out,
    const float* __restrict__ bias,
    const float* __restrict__ bnG, const float* __restrict__ bnB,
    const float* __restrict__ bnM, const float* __restrict__ bnV,
    const float* __restrict__ resPtr, int doRelu,
    int M, int Nout, int numKT,
    int csLog2, int Tin, int Tout, int stride)
{
    __shared__ unsigned short Bs[128][32];

    const int tid  = threadIdx.x;
    const int lane = tid & 31;
    const int wave = tid >> 5;
    const int wm   = wave & 3;
    const int wn   = wave >> 2;
    const int lr   = lane & 15;
    const int hs   = lane >> 4;

    const int n0 = blockIdx.x * 128;
    const int mt = blockIdx.y * 4 + wm;
    const int C  = 1 << csLog2;

    v8f acc[4];
#pragma unroll
    for (int j = 0; j < 4; j++)
#pragma unroll
        for (int e = 0; e < 8; e++) acc[j][e] = 0.0f;

    // staging role: each thread owns one (n-row, 16-channel half) = 32B
    const int nl   = tid & 127;
    const int part = tid >> 7;            // 0 or 1 -> channels part*16..+15
    // hoisted K-invariant decomposition of n
    const int n  = n0 + nl;
    const int vv = n % 17;
    const int r2 = n / 17;
    const int to = r2 % Tout;
    const int bb = r2 / Tout;
    const int toS     = to * stride;
    const int rowBase = bb * Tin;

    // 32-bit LDS offset of this thread's staging slot
    const unsigned ldsAddr =
        (unsigned)(size_t)(&Bs[nl][part * 16]);

    const unsigned short* aBase = Aswz + ((size_t)mt * numKT) * 512 + (size_t)lane * 8;

    for (int kt = 0; kt < numKT; kt++) {
        const int k0    = kt * 32;
        const int dt    = k0 >> csLog2;          // whole tile in one tap
        const int cbase = k0 & (C - 1);
        const int t     = toS + dt - 4;
        const bool ok   = (unsigned)t < (unsigned)Tin;
        const unsigned short* src = ok
            ? yBF + (size_t)((rowBase + t) * 17 + vv) * C + cbase + part * 16
            : zeroPage;
        // async copy 32B global -> LDS (offset applies to both addresses)
        asm volatile("global_load_async_to_lds_b128 %0, %1, off"
                     :: "v"(ldsAddr), "v"(src) : "memory");
        asm volatile("global_load_async_to_lds_b128 %0, %1, off offset:16"
                     :: "v"(ldsAddr), "v"(src) : "memory");
        asm volatile("s_wait_asynccnt 0" ::: "memory");
        __syncthreads();

        v8bfv alo = *reinterpret_cast<const v8bfv*>(aBase + (size_t)kt * 512);
        v8bfv ahi = *reinterpret_cast<const v8bfv*>(aBase + (size_t)kt * 512 + 256);
        v16bf a = frag16(alo, ahi);
#pragma unroll
        for (int j = 0; j < 4; j++) {
            int nb = wn * 64 + j * 16;
            v8bfv blo = *reinterpret_cast<const v8bfv*>(&Bs[nb + lr][hs * 8]);
            v8bfv bhi = *reinterpret_cast<const v8bfv*>(&Bs[nb + lr][16 + hs * 8]);
            v16bf b = frag16(blo, bhi);
            acc[j] = __builtin_amdgcn_wmma_f32_16x16x32_bf16(
                false, a, false, b, (short)0, acc[j], false, false);
        }
        __syncthreads();
    }

#pragma unroll
    for (int j = 0; j < 4; j++) {
#pragma unroll
        for (int r = 0; r < 8; r++) {
            int m = mt * 16 + r + 8 * hs;
            int nn = n0 + wn * 64 + j * 16 + lr;
            float v = ep_val(acc[j][r], m, nn, bias, 1, M,
                             bnG, bnB, bnM, bnV, resPtr, doRelu,
                             nullptr, nullptr, nullptr, nullptr, Nout);
            Out[(size_t)m * Nout + nn] = v;
        }
    }
}

// ---------------------------------------------------------------------------
// data_bn + layout transform: x(B,C,T,V) -> act[c][(b*T+t)*V+v], params idx v*C+c
// ---------------------------------------------------------------------------
__global__ void data_bn_kernel(const float* __restrict__ x,
                               const float* __restrict__ gamma, const float* __restrict__ beta,
                               const float* __restrict__ mean,  const float* __restrict__ var,
                               float* __restrict__ out, int Bn, int C, int T, int V)
{
    size_t i = (size_t)blockIdx.x * blockDim.x + threadIdx.x;
    size_t N = (size_t)Bn * T * V;
    size_t total = (size_t)C * N;
    if (i >= total) return;
    int c = (int)(i / N);
    size_t n = i - (size_t)c * N;
    int v = (int)(n % V);
    size_t r = n / V;
    int t = (int)(r % T);
    int b = (int)(r / T);
    float xv = x[(((size_t)b * C + c) * T + t) * V + v];
    int p = v * C + c;
    out[i] = (xv - mean[p]) * (gamma[p] * rsqrtf(var[p] + 1e-5f)) + beta[p];
}

// Ak = A * edge + adaptive  (3 x 17 x 17)
__global__ void ak_kernel(const float* __restrict__ A, const float* __restrict__ edge,
                          const float* __restrict__ adaptive, float* __restrict__ Ak, int n)
{
    int i = blockIdx.x * blockDim.x + threadIdx.x;
    if (i >= n) return;
    Ak[i] = A[i] * edge[i] + adaptive[i % 289];
}

// zero-fill helper
__global__ void zfill_kernel(unsigned short* __restrict__ p, int n)
{
    int i = blockIdx.x * blockDim.x + threadIdx.x;
    if (i < n) p[i] = 0;
}

// Graph mixing directly into the swizzled B fragment layout.
__global__ void mix_swz_kernel(const float* __restrict__ act, const float* __restrict__ Ak,
                               unsigned short* __restrict__ out,
                               int C, int K3, int numKT, int Nn)
{
    __shared__ float sAk[3 * 17 * 17];
    for (int i = threadIdx.x; i < 3 * 17 * 17; i += blockDim.x) sAk[i] = Ak[i];
    __syncthreads();
    size_t i = (size_t)blockIdx.x * blockDim.x + threadIdx.x;
    size_t total = (size_t)(Nn / 16) * numKT * 512;
    if (i >= total) return;
    size_t blk = i >> 9;
    int within = (int)(i & 511);
    int h = within >> 8;
    int le = within & 255;
    int lane = le >> 3;
    int e = le & 7;
    int nt = (int)(blk / numKT);
    int kt = (int)(blk - (size_t)nt * numKT);
    int n = nt * 16 + (lane & 15);
    int k = kt * 32 + h * 16 + (lane >> 4) * 8 + e;
    unsigned short outv = 0;
    if (k < K3) {
        int kg = k / C, c = k - kg * C;
        int w = n % 17;
        size_t base = (size_t)n - w;
        const float* arow = act + (size_t)c * Nn + base;
        const float* ak = sAk + kg * 289;
        float s = 0.f;
#pragma unroll
        for (int v = 0; v < 17; v++) s += arow[v] * ak[v * 17 + w];
        outv = f2bf(s);
    }
    out[i] = outv;
}

// Pack weights (fp32) into the swizzled A fragment layout, zero-padded in K.
// mode 0: 1x1 w(O,Cin)   -> A[o][c]        = w[o*Cin+c]
// mode 1: gcn w(3,O,Cin) -> A[o][kg*Cin+c] = w[(kg*M+o)*Cin+c]
// mode 2: tcn w(O,Cin,9) -> A[o][dt*Cin+c] = w[(o*Cin+c)*9+dt]
__global__ void pack_w_swz_kernel(const float* __restrict__ w, unsigned short* __restrict__ out,
                                  int M, int K, int numKT, int mode, int Cin)
{
    size_t i = (size_t)blockIdx.x * blockDim.x + threadIdx.x;
    size_t total = (size_t)(M / 16) * numKT * 512;
    if (i >= total) return;
    size_t blk = i >> 9;
    int within = (int)(i & 511);
    int h = within >> 8;
    int le = within & 255;
    int lane = le >> 3;
    int e = le & 7;
    int mt = (int)(blk / numKT);
    int kt = (int)(blk - (size_t)mt * numKT);
    int m = mt * 16 + (lane & 15);
    int k = kt * 32 + h * 16 + (lane >> 4) * 8 + e;
    float f = 0.f;
    if (k < K) {
        if (mode == 0) {
            f = w[(size_t)m * Cin + k];
        } else if (mode == 1) {
            int kg = k / Cin, c = k - kg * Cin;
            f = w[((size_t)kg * M + m) * Cin + c];
        } else {
            int dt = k / Cin, c = k - dt * Cin;
            f = w[((size_t)m * Cin + c) * 9 + dt];
        }
    }
    out[i] = f2bf(f);
}

// s[b][c*V+v] = mean_t x[c][(b*T+t)*V+v]
__global__ void pool_t_kernel(const float* __restrict__ x, float* __restrict__ s,
                              int Bn, int C, int T, int V)
{
    int i = blockIdx.x * blockDim.x + threadIdx.x;
    if (i >= Bn * C * V) return;
    int v = i % V; int r = i / V; int c = r % C; int b = r / C;
    size_t N = (size_t)Bn * T * V;
    const float* p = x + (size_t)c * N + (size_t)b * T * V + v;
    float acc = 0.f;
    for (int t = 0; t < T; t++) acc += p[(size_t)t * V];
    s[(size_t)b * C * V + (size_t)c * V + v] = acc / (float)T;
}

// out[b][o] = act( in[b][:] . w[o][:] + bias[o] );  act: 0 none, 1 relu, 2 sigmoid
__global__ void fc_kernel(const float* __restrict__ in, const float* __restrict__ w,
                          const float* __restrict__ bias, float* __restrict__ out,
                          int Bn, int In, int On, int act)
{
    int i = blockIdx.x * blockDim.x + threadIdx.x;
    if (i >= Bn * On) return;
    int o = i % On, b = i / On;
    const float* ip = in + (size_t)b * In;
    const float* wp = w + (size_t)o * In;
    float acc = bias[o];
    for (int j = 0; j < In; j++) acc += ip[j] * wp[j];
    if (act == 1) acc = fmaxf(acc, 0.f);
    else if (act == 2) acc = 1.f / (1.f + __expf(-acc));
    out[i] = acc;
}

// tm[b][c][t] = (1/V) sum_v x[c][(b*T+t)*V+v] * sa[b*V+v]
__global__ void tmean_kernel(const float* __restrict__ x, const float* __restrict__ sa,
                             float* __restrict__ tm, int Bn, int C, int T, int V)
{
    int i = blockIdx.x * blockDim.x + threadIdx.x;
    if (i >= Bn * C * T) return;
    int t = i % T; int r = i / T; int c = r % C; int b = r / C;
    size_t N = (size_t)Bn * T * V;
    const float* p = x + (size_t)c * N + ((size_t)b * T + t) * V;
    const float* s = sa + (size_t)b * V;
    float acc = 0.f;
#pragma unroll
    for (int v = 0; v < 17; v++) acc += p[v] * s[v];
    tm[i] = acc / (float)V;
}

// h2[b][m][t] = relu( sum_c w[m*C+c] * tm[b][c][t] + bias[m] )
__global__ void t1_kernel(const float* __restrict__ tm, const float* __restrict__ w,
                          const float* __restrict__ bias, float* __restrict__ h2,
                          int Bn, int C, int T, int Mid)
{
    int i = blockIdx.x * blockDim.x + threadIdx.x;
    if (i >= Bn * Mid * T) return;
    int t = i % T; int r = i / T; int m = r % Mid; int b = r / Mid;
    const float* tp = tm + (size_t)b * C * T + t;
    const float* wp = w + (size_t)m * C;
    float acc = bias[m];
    for (int c = 0; c < C; c++) acc += wp[c] * tp[(size_t)c * T];
    h2[i] = fmaxf(acc, 0.f);
}

// ta[b][t] = sigmoid( sum_m w[m] * h2[b][m][t] + bias[0] )
__global__ void t2_kernel(const float* __restrict__ h2, const float* __restrict__ w,
                          const float* __restrict__ bias, float* __restrict__ ta,
                          int Bn, int Mid, int T)
{
    int i = blockIdx.x * blockDim.x + threadIdx.x;
    if (i >= Bn * T) return;
    int t = i % T; int b = i / T;
    const float* hp = h2 + (size_t)b * Mid * T + t;
    float acc = bias[0];
    for (int m = 0; m < Mid; m++) acc += w[m] * hp[(size_t)m * T];
    ta[i] = 1.f / (1.f + __expf(-acc));
}

// g[b][c] = (1/(T*V)) sum_t ta[b*T+t] * sum_v x[c][(b*T+t)*V+v] * sa[b*V+v]
__global__ void gpool_kernel(const float* __restrict__ x, const float* __restrict__ sa,
                             const float* __restrict__ ta, float* __restrict__ g,
                             int Bn, int C, int T, int V)
{
    int i = blockIdx.x * blockDim.x + threadIdx.x;
    if (i >= Bn * C) return;
    int c = i % C; int b = i / C;
    size_t N = (size_t)Bn * T * V;
    const float* sp = sa + (size_t)b * V;
    float acc = 0.f;
    for (int t = 0; t < T; t++) {
        const float* p = x + (size_t)c * N + ((size_t)b * T + t) * V;
        float inner = 0.f;
#pragma unroll
        for (int v = 0; v < 17; v++) inner += p[v] * sp[v];
        acc += ta[(size_t)b * T + t] * inner;
    }
    g[i] = acc / (float)(T * V);
}

// ---------------------------------------------------------------------------
// Host orchestration
// ---------------------------------------------------------------------------
extern "C" void kernel_launch(void* const* d_in, const int* in_sizes, int n_in,
                              void* d_out, int out_size, void* d_ws, size_t ws_size,
                              hipStream_t stream)
{
    (void)in_sizes; (void)n_in; (void)out_size; (void)ws_size;

    const int Bc = 128, Tc = 256, Vc = 17, NC = 60, MID = 64;
    struct BlkCfg { int cin, cout, stride; };
    const BlkCfg cfg[6] = {{9,64,1},{64,64,1},{64,128,2},{128,128,1},{128,256,2},{256,256,1}};

    // ---- param index map: jax pytree flatten order (sorted dict keys, None dropped)
    int idx = 2;  // d_in[0]=x, d_in[1]=A
    int AT_fc1_b = idx++, AT_fc1_w = idx++, AT_fc2_b = idx++, AT_fc2_w = idx++;
    int AT_t1_b  = idx++, AT_t1_w  = idx++, AT_t2_b  = idx++, AT_t2_w  = idx++;
    int G_adapt[6], G_bnB[6], G_bnG[6], G_bnM[6], G_bnV[6], G_cb[6], G_cw[6], G_edge[6];
    int Gr_bnB[6], Gr_bnG[6], Gr_bnM[6], Gr_bnV[6], Gr_cb[6], Gr_cw[6];
    int Br_bnB[6], Br_bnG[6], Br_bnM[6], Br_bnV[6], Br_cb[6], Br_cw[6];
    int T_bn1B[6], T_bn1G[6], T_bn1M[6], T_bn1V[6];
    int T_bn2B[6], T_bn2G[6], T_bn2M[6], T_bn2V[6], T_cb[6], T_cw[6];
    for (int i = 0; i < 6; i++) {
        bool gres = cfg[i].cin != cfg[i].cout;
        bool bres = !(cfg[i].stride == 1 && cfg[i].cin == cfg[i].cout);
        G_adapt[i] = idx++;
        G_bnB[i] = idx++; G_bnG[i] = idx++; G_bnM[i] = idx++; G_bnV[i] = idx++;
        G_cb[i]  = idx++; G_cw[i]  = idx++;
        G_edge[i] = idx++;
        if (gres) {
            Gr_bnB[i] = idx++; Gr_bnG[i] = idx++; Gr_bnM[i] = idx++; Gr_bnV[i] = idx++;
            Gr_cb[i]  = idx++; Gr_cw[i]  = idx++;
        } else { Gr_bnB[i]=Gr_bnG[i]=Gr_bnM[i]=Gr_bnV[i]=Gr_cb[i]=Gr_cw[i] = -1; }
        if (bres) {
            Br_bnB[i] = idx++; Br_bnG[i] = idx++; Br_bnM[i] = idx++; Br_bnV[i] = idx++;
            Br_cb[i]  = idx++; Br_cw[i]  = idx++;
        } else { Br_bnB[i]=Br_bnG[i]=Br_bnM[i]=Br_bnV[i]=Br_cb[i]=Br_cw[i] = -1; }
        T_bn1B[i] = idx++; T_bn1G[i] = idx++; T_bn1M[i] = idx++; T_bn1V[i] = idx++;
        T_bn2B[i] = idx++; T_bn2G[i] = idx++; T_bn2M[i] = idx++; T_bn2V[i] = idx++;
        T_cb[i]   = idx++; T_cw[i]   = idx++;
    }
    int DB_beta = idx++, DB_gamma = idx++, DB_mean = idx++, DB_var = idx++;
    int FC_b = idx++, FC_w = idx++;

    auto F = [&](int i) -> const float* { return (const float*)d_in[i]; };
    const float* dx = F(0);
    const float* dA = F(1);

    // ---- workspace carve-out ----
    char* ws = (char*)d_ws;
    size_t off = 0;
    auto walloc = [&](size_t bytes) -> void* {
        void* p = ws + off; off += (bytes + 255) & ~(size_t)255; return p;
    };
    const size_t NMAX = (size_t)64 * 557056;
    float* P0 = (float*)walloc(NMAX * 4);
    float* P1 = (float*)walloc(NMAX * 4);
    unsigned short* Yb = (unsigned short*)walloc(NMAX * 2);                    // y bf16 [n][C]
    float* Rb = (float*)walloc(NMAX * 4);
    unsigned short* XA = (unsigned short*)walloc((size_t)3 * 64 * 557056 * 2); // swizzled B
    unsigned short* WB = (unsigned short*)walloc((size_t)256 * 2304 * 2);      // swizzled A
    unsigned short* ZP = (unsigned short*)walloc(64 * 2);                      // zero page
    float* AK  = (float*)walloc(867 * 4);
    float* Sb  = (float*)walloc((size_t)Bc * 256 * Vc * 4);
    float* H1  = (float*)walloc((size_t)Bc * MID * 4);
    float* SA  = (float*)walloc((size_t)Bc * Vc * 4);
    float* TM  = (float*)walloc((size_t)Bc * 256 * 64 * 4);
    float* H2  = (float*)walloc((size_t)Bc * MID * 64 * 4);
    float* TAb = (float*)walloc((size_t)Bc * 64 * 4);
    float* Gb  = (float*)walloc((size_t)Bc * 256 * 4);

    auto cdiv = [](size_t a, size_t b) { return (unsigned)((a + b - 1) / b); };
    auto ilog2 = [](int v) { int l = 0; while ((1 << l) < v) l++; return l; };

    // zero page for async OOB redirect
    zfill_kernel<<<1, 64, 0, stream>>>(ZP, 64);

    // ---- data_bn: x -> P0 in [C][N] layout ----
    {
        size_t total = (size_t)9 * Bc * Tc * Vc;
        data_bn_kernel<<<cdiv(total, 256), 256, 0, stream>>>(
            dx, F(DB_gamma), F(DB_beta), F(DB_mean), F(DB_var), P0, Bc, 9, Tc, Vc);
    }

    float* cur = P0; float* nxt = P1;
    int Tin = Tc;

    for (int i = 0; i < 6; i++) {
        const int cin = cfg[i].cin, cout = cfg[i].cout, s = cfg[i].stride;
        const int Tout = Tin / s;
        const int Nin  = Bc * Tin * Vc;
        const int Nout = Bc * Tout * Vc;
        const bool gres = cin != cout;
        const bool bres = !(s == 1 && cin == cout);
        const int K3 = 3 * cin;
        const int nKT3 = (K3 + 31) / 32;
        const int nKTr = (cin + 31) / 32;
        const int nKTt = (9 * cout) / 32;   // exact: 9*cout % 32 == 0

        // Ak = A*edge + adaptive
        ak_kernel<<<cdiv(867, 256), 256, 0, stream>>>(dA, F(G_edge[i]), F(G_adapt[i]), AK, 867);

        // graph mixing -> swizzled bf16 B operand
        {
            size_t total = (size_t)(Nin / 16) * nKT3 * 512;
            mix_swz_kernel<<<cdiv(total, 256), 256, 0, stream>>>(cur, AK, XA, cin, K3, nKT3, Nin);
        }

        // GCN residual path (1x1 + BN), if present
        const float* gresPtr = cur;
        if (gres) {
            size_t wt = (size_t)(cout / 16) * nKTr * 512;
            pack_w_swz_kernel<<<cdiv(wt, 256), 256, 0, stream>>>(
                F(Gr_cw[i]), WB, cout, cin, nKTr, 0, cin);
            wmma_gemm_1x1_kernel<<<dim3((unsigned)(Nin / 128), cout / 64), 256, 0, stream>>>(
                WB, cur, Rb, F(Gr_cb[i]), 1,
                F(Gr_bnG[i]), F(Gr_bnB[i]), F(Gr_bnM[i]), F(Gr_bnV[i]),
                cout, cin, Nin, Nin, nKTr, Tin, Tin, 1);
            gresPtr = Rb;
        }

        // GCN main GEMM (swizzled x swizzled, no LDS) -> y bf16 [n][cout]:
        // y = relu(BN_tcn1(relu(BN_gcn(W*xA + sum_k b) + res)))
        {
            size_t wt = (size_t)(cout / 16) * nKT3 * 512;
            pack_w_swz_kernel<<<cdiv(wt, 256), 256, 0, stream>>>(
                F(G_cw[i]), WB, cout, K3, nKT3, 1, cin);
            wmma_gemm_swz_kernel<<<dim3((unsigned)(Nin / 128), cout / 64), 256, 0, stream>>>(
                WB, XA, Yb, F(G_cb[i]), 2,
                F(G_bnG[i]), F(G_bnB[i]), F(G_bnM[i]), F(G_bnV[i]),
                gresPtr, 1,
                F(T_bn1G[i]), F(T_bn1B[i]), F(T_bn1M[i]), F(T_bn1V[i]),
                cout, Nin, nKT3);
        }

        // Block residual path (strided 1x1 + BN), if present
        const float* bresPtr = cur;
        if (bres) {
            size_t wt = (size_t)(cout / 16) * nKTr * 512;
            pack_w_swz_kernel<<<cdiv(wt, 256), 256, 0, stream>>>(
                F(Br_cw[i]), WB, cout, cin, nKTr, 0, cin);
            wmma_gemm_1x1_kernel<<<dim3((unsigned)(Nout / 128), cout / 64), 256, 0, stream>>>(
                WB, cur, Rb, F(Br_cb[i]), 1,
                F(Br_bnG[i]), F(Br_bnB[i]), F(Br_bnM[i]), F(Br_bnV[i]),
                cout, cin, Nout, Nin, nKTr, Tin, Tout, s);
            bresPtr = Rb;
        }

        // TCN implicit GEMM (9-tap temporal conv, stride s) with async staging:
        // out = relu(BN_tcn2(conv(y) + b) + res)
        {
            size_t wt = (size_t)(cout / 16) * nKTt * 512;
            pack_w_swz_kernel<<<cdiv(wt, 256), 256, 0, stream>>>(
                F(T_cw[i]), WB, cout, 9 * cout, nKTt, 2, cout);
            wmma_gemm_tconv_kernel<<<dim3((unsigned)(Nout / 128), cout / 64), 256, 0, stream>>>(
                WB, Yb, ZP, nxt, F(T_cb[i]),
                F(T_bn2G[i]), F(T_bn2B[i]), F(T_bn2M[i]), F(T_bn2V[i]),
                bresPtr, 1,
                cout, Nout, nKTt, ilog2(cout), Tin, Tout, s);
        }

        float* tmp = cur; cur = nxt; nxt = tmp;
        Tin = Tout;
    }

    // ---- attention + classifier tail (scalar; negligible FLOPs) ----
    const int T6 = Tin;   // 64
    const int C6 = 256;
    pool_t_kernel<<<cdiv((size_t)Bc * C6 * Vc, 256), 256, 0, stream>>>(cur, Sb, Bc, C6, T6, Vc);
    fc_kernel<<<cdiv((size_t)Bc * MID, 256), 256, 0, stream>>>(
        Sb, F(AT_fc1_w), F(AT_fc1_b), H1, Bc, C6 * Vc, MID, 1);
    fc_kernel<<<cdiv((size_t)Bc * Vc, 256), 256, 0, stream>>>(
        H1, F(AT_fc2_w), F(AT_fc2_b), SA, Bc, MID, Vc, 2);
    tmean_kernel<<<cdiv((size_t)Bc * C6 * T6, 256), 256, 0, stream>>>(cur, SA, TM, Bc, C6, T6, Vc);
    t1_kernel<<<cdiv((size_t)Bc * MID * T6, 256), 256, 0, stream>>>(
        TM, F(AT_t1_w), F(AT_t1_b), H2, Bc, C6, T6, MID);
    t2_kernel<<<cdiv((size_t)Bc * T6, 256), 256, 0, stream>>>(
        H2, F(AT_t2_w), F(AT_t2_b), TAb, Bc, MID, T6);
    gpool_kernel<<<cdiv((size_t)Bc * C6, 256), 256, 0, stream>>>(cur, SA, TAb, Gb, Bc, C6, T6, Vc);
    fc_kernel<<<cdiv((size_t)Bc * NC, 256), 256, 0, stream>>>(
        Gb, F(FC_w), F(FC_b), (float*)d_out, Bc, C6, NC, 0);
}
